// PredictorLG_58540404244844
// MI455X (gfx1250) — matrix operand order repaired and checked by reading, not compiled, not run on hardware
//
#include <hip/hip_runtime.h>
#include <hip/hip_bf16.h>

typedef __attribute__((ext_vector_type(16))) _Float16 v16h;
typedef __attribute__((ext_vector_type(8)))  float    v8f;
typedef unsigned int u32;

#define CHW      2304      // 48*48
#define C_DIM    768
#define CH_DIM   192
#define NT_TILES 12        // 192 / 16
#define KSTEPS   24        // 768 / 32
#define XPAD     769       // odd stride -> conflict-free LDS rows
#define TOKS     32        // tokens per block (2 M-tiles)
#define NWAVES   6
#define NTHREADS 192
#define W1S_DWORDS_PER_M (NT_TILES*KSTEPS*32*8)   // 73728 dwords / modality

// ---------------------------------------------------------------------------
// Prep: swizzle w1 (f32 [3,768,192]) into f16 WMMA-B fragment layout.
// Chunk (m, nt, ks) is 1KB: [lane 0..31][vgpr i 0..7] packed f16 pairs so a
// wave's B-fragment load is lane*32B contiguous (coalesced global_load_b128).
// B k order used here (contiguous, same as A): VGPR i, lanes 0-15: K=2i,2i+1 ; lanes 16-31: K=16+2i,17+2i
// ---------------------------------------------------------------------------
__global__ void prep_w1_swizzle(const float* __restrict__ w1, u32* __restrict__ w1s) {
    int idx = blockIdx.x * 256 + threadIdx.x;            // 3*73728 = 221184 total
    if (idx >= 3 * W1S_DWORDS_PER_M) return;
    int i    = idx & 7;
    int lane = (idx >> 3) & 31;
    int ks   = (idx >> 8) % KSTEPS;
    int nt   = ((idx >> 8) / KSTEPS) % NT_TILES;
    int m    = (idx >> 8) / (KSTEPS * NT_TILES);
    int n     = nt * 16 + (lane & 15);
    int khalf = lane >> 4;
    int k     = ks * 32 + khalf * 16 + 2 * i;
    float v0 = w1[((size_t)m * C_DIM + k)     * CH_DIM + n];
    float v1 = w1[((size_t)m * C_DIM + k + 1) * CH_DIM + n];
    union { _Float16 h[2]; u32 u; } pk;
    pk.h[0] = (_Float16)v0; pk.h[1] = (_Float16)v1;
    w1s[idx] = pk.u;
}

// ---------------------------------------------------------------------------
// Main kernel: one block = 32 tokens of one (modality, batch).
// ---------------------------------------------------------------------------
__global__ __launch_bounds__(NTHREADS)
void predictor_lg_kernel(const float* __restrict__ f0, const float* __restrict__ f1,
                         const float* __restrict__ f2,
                         const float* __restrict__ ln_g, const float* __restrict__ ln_b,
                         const float* __restrict__ b1,   const float* __restrict__ b2,
                         const float* __restrict__ w2,
                         const u32*   __restrict__ w1s,
                         float* __restrict__ out) {
    __shared__ float Xs[TOKS * XPAD];                       // 98.4 KB f32 token stage
    __shared__ __align__(32) u32 A_u[2 * KSTEPS * 32 * 8];  // 48 KB f16 A fragments
    __shared__ float Gs[C_DIM], Bs[C_DIM];
    __shared__ float score[TOKS];

    const int tid  = threadIdx.x;
    const int lane = tid & 31;
    const int wave = tid >> 5;

    const int blocksPerM = 16 * (CHW / TOKS);               // 1152
    const int m   = blockIdx.x / blocksPerM;
    const int rem = blockIdx.x % blocksPerM;
    const int bb  = rem / (CHW / TOKS);
    const int hw0 = (rem % (CHW / TOKS)) * TOKS;

    const float* feat = (m == 0 ? f0 : (m == 1 ? f1 : f2))
                        + (size_t)bb * C_DIM * CHW + hw0;

    // ---- Phase 0: async staging (CDNA5 global->LDS direct path) ---------
    if (tid < TOKS) score[tid] = 0.0f;
    for (int c = tid; c < C_DIM; c += NTHREADS) {
        Gs[c] = ln_g[m * C_DIM + c];
        Bs[c] = ln_b[m * C_DIM + c];
    }
    {
        // lane = token (contiguous hw -> 128B/wave coalesced), waves stride channels
        const u32 ldsb = (u32)(uintptr_t)(&Xs[0]);
        u32 dso = ldsb + (((u32)lane * XPAD + (u32)wave) << 2);   // Xs[lane][c]
        u32 gvo = (((u32)wave * CHW + (u32)lane) << 2);           // feat[c][lane]
        for (int c = wave; c < C_DIM; c += NWAVES) {
            asm volatile("global_load_async_to_lds_b32 %0, %1, %2"
                         :: "v"(dso), "v"(gvo), "s"(feat) : "memory");
            dso += (u32)(NWAVES * 4);
            gvo += (u32)(NWAVES * CHW * 4);
        }
        asm volatile("s_wait_asynccnt 0x0" ::: "memory");
    }
    __syncthreads();

    // ---- Phase 1: LayerNorm -> f16 A fragments --------------------------
    // A k order used here (contiguous, same as B), pair p=K>>1: khalf=p>>3, vgpr=p&7,
    // fragment lane = M + 16*khalf.
    for (int tok = wave; tok < TOKS; tok += NWAVES) {
        const float* xr = &Xs[tok * XPAD];
        float s = 0.0f, q = 0.0f;
        #pragma unroll
        for (int j = 0; j < 24; ++j) {
            float v = xr[lane + 32 * j];
            s += v; q += v * v;
        }
        #pragma unroll
        for (int off = 16; off; off >>= 1) {
            s += __shfl_xor(s, off);
            q += __shfl_xor(q, off);
        }
        const float mu  = s * (1.0f / 768.0f);
        const float var = q * (1.0f / 768.0f) - mu * mu;
        const float rs  = rsqrtf(var + 1e-5f);

        const int mt = tok >> 4, mrow = tok & 15;
        const int p  = lane & 15;
        const int khalf = p >> 3;
        const int vg    = p & 7;
        const int fraglane = mrow + 16 * khalf;
        #pragma unroll
        for (int j = 0; j < 12; ++j) {
            int c0 = 2 * lane + 64 * j;          // consecutive K pair (c0, c0+1)
            int ks = c0 >> 5;
            float a0 = (xr[c0]     - mu) * rs * Gs[c0]     + Bs[c0];
            float a1 = (xr[c0 + 1] - mu) * rs * Gs[c0 + 1] + Bs[c0 + 1];
            union { _Float16 h[2]; u32 u; } pk;
            pk.h[0] = (_Float16)a0; pk.h[1] = (_Float16)a1;
            A_u[(mt * KSTEPS + ks) * 256 + fraglane * 8 + vg] = pk.u;
        }
    }
    __syncthreads();

    // ---- Phase 2: WMMA GEMM  [32 x 768] x [768 x 192] -------------------
    // Single per-lane byte pointer; fully unrolled loop -> all B loads are
    // global_load_b128 with immediate offsets (no address VALU in the loop).
    v8f acc00 = {}, acc01 = {}, acc10 = {}, acc11 = {};
    const int nt0 = wave * 2;                    // wave owns 2 N-tiles
    const char* bbase = (const char*)(w1s + (size_t)m * W1S_DWORDS_PER_M
                                          + (size_t)nt0 * KSTEPS * 256)
                        + (size_t)lane * 32;
    const char* abase = (const char*)(&A_u[0]) + (size_t)lane * 32;
    #pragma unroll
    for (int ks = 0; ks < KSTEPS; ++ks) {
        v16h a0  = *(const v16h*)(abase + ks * 1024);                    // M-tile 0
        v16h a1  = *(const v16h*)(abase + (KSTEPS + ks) * 1024);         // M-tile 1
        v16h bf0 = *(const v16h*)(bbase + ks * 1024);                    // N-tile nt0
        v16h bf1 = *(const v16h*)(bbase + (KSTEPS + ks) * 1024);         // N-tile nt0+1
        acc00 = __builtin_amdgcn_wmma_f32_16x16x32_f16(false, a0, false, bf0, (short)0, acc00, false, false);
        acc01 = __builtin_amdgcn_wmma_f32_16x16x32_f16(false, a0, false, bf1, (short)0, acc01, false, false);
        acc10 = __builtin_amdgcn_wmma_f32_16x16x32_f16(false, a1, false, bf0, (short)0, acc10, false, false);
        acc11 = __builtin_amdgcn_wmma_f32_16x16x32_f16(false, a1, false, bf1, (short)0, acc11, false, false);
    }

    // ---- Epilogue: +B1, exact GELU, x w2[n], reduce over n --------------
    const int mhalf = lane >> 4;
    #pragma unroll
    for (int ntl = 0; ntl < 2; ++ntl) {
        const int n = (nt0 + ntl) * 16 + (lane & 15);
        const float bias = b1[m * CH_DIM + n];
        const float wn   = w2[m * CH_DIM + n];
        #pragma unroll
        for (int mt = 0; mt < 2; ++mt) {
            v8f cc = (mt == 0) ? (ntl == 0 ? acc00 : acc01)
                               : (ntl == 0 ? acc10 : acc11);
            #pragma unroll
            for (int i = 0; i < 8; ++i) {
                float h  = cc[i] + bias;
                float ge = 0.5f * h * (1.0f + erff(h * 0.70710678118654752f));
                float val = ge * wn;
                #pragma unroll
                for (int off = 8; off; off >>= 1)   // reduce 16 lanes / half
                    val += __shfl_xor(val, off);
                if ((lane & 15) == 0)
                    atomicAdd(&score[mt * 16 + mhalf * 8 + i], val);
            }
        }
    }
    __syncthreads();

    // ---- Sigmoid + store ------------------------------------------------
    if (tid < TOKS) {
        float x = score[tid] + b2[m];
        float sg = 1.0f / (1.0f + __expf(-x));
        out[(size_t)m * 16 * CHW + (size_t)bb * CHW + hw0 + tid] = sg;
    }
}

// ---------------------------------------------------------------------------
// Inputs: 0=feat0 1=feat1 2=feat2 3=ln_g 4=ln_b 5=w1 6=b1 7=w2 8=b2
// ---------------------------------------------------------------------------
extern "C" void kernel_launch(void* const* d_in, const int* in_sizes, int n_in,
                              void* d_out, int out_size, void* d_ws, size_t ws_size,
                              hipStream_t stream) {
    const float* f0   = (const float*)d_in[0];
    const float* f1   = (const float*)d_in[1];
    const float* f2   = (const float*)d_in[2];
    const float* ln_g = (const float*)d_in[3];
    const float* ln_b = (const float*)d_in[4];
    const float* w1   = (const float*)d_in[5];
    const float* b1   = (const float*)d_in[6];
    const float* w2   = (const float*)d_in[7];
    const float* b2   = (const float*)d_in[8];
    u32* w1s = (u32*)d_ws;                       // 884,736 bytes used

    // swizzle weights every call (idempotent, deterministic)
    prep_w1_swizzle<<<(3 * W1S_DWORDS_PER_M + 255) / 256, 256, 0, stream>>>(w1, w1s);

    const int blocks = 3 * 16 * (CHW / TOKS);    // 3456
    predictor_lg_kernel<<<blocks, NTHREADS, 0, stream>>>(
        f0, f1, f2, ln_g, ln_b, b1, b2, w2, w1s, (float*)d_out);
}
